// Node_Convolution_7499012898889
// MI455X (gfx1250) — matrix operand order repaired and checked by reading, not compiled, not run on hardware
//
#include <hip/hip_runtime.h>
#include <hip/hip_bf16.h>
#include <math.h>

typedef __attribute__((ext_vector_type(16))) __bf16 v16bf;
typedef __attribute__((ext_vector_type(8)))  float  v8f;
typedef int v4i __attribute__((vector_size(16)));

#define D        128
#define KTOT     384      // 3*D
#define KC       12       // K chunks of 32
#define NC       8        // N chunks of 16
#define TILES    2        // 16-edge tiles per block
#define ZSTRIDE  392      // padded LDS row stride (bf16 elems); 392*2 = 784 B (16B aligned)
#define PACK_HALF 49152   // elems per packed matrix: 12*8*32*16

#define AS1 __attribute__((address_space(1)))
#define AS3 __attribute__((address_space(3)))

#if defined(__has_builtin)
#if __has_builtin(__builtin_amdgcn_global_load_async_to_lds_b128)
#define USE_ASYNC_LDS 1
#endif
#endif

__device__ __forceinline__ unsigned short f2bf(float f) {
    __bf16 h = (__bf16)f;
    return __builtin_bit_cast(unsigned short, h);
}
__device__ __forceinline__ float sigmoid_(float x)  { return 1.0f / (1.0f + __expf(-x)); }
__device__ __forceinline__ float softplus_(float x) { return x > 15.0f ? x : log1pf(__expf(x)); }

__device__ __forceinline__ void wait_async_zero() {
#if defined(__has_builtin)
#if __has_builtin(__builtin_amdgcn_s_wait_asynccnt)
    __builtin_amdgcn_s_wait_asynccnt(0);
    return;
#else
    asm volatile("s_wait_asynccnt 0x0" ::: "memory");
    return;
#endif
#endif
}

// ---------------------------------------------------------------- zero scratch
__global__ void k_zero(float* __restrict__ p, int n) {
    int i = blockIdx.x * 256 + threadIdx.x;
    if (i < n) p[i] = 0.0f;
}

// ----------------------------------------------------- pre-convert x -> bf16
__global__ void k_cvt_x(const float* __restrict__ x, unsigned short* __restrict__ xbf, int n) {
    int i = blockIdx.x * 256 + threadIdx.x;
    if (i < n) xbf[i] = f2bf(x[i]);
}

// ------------------------------------------------- pack Wf/Ws -> bf16 B frags
// B fragment (bf16 32x16, K x N): lane l holds 16 consecutive bf16:
//   k = kc*32 + (l<16 ? 0 : 16) + e,  n = nc*16 + (l&15)
// packed linear index = ((sel*96 + kc*8 + nc)*32 + l)*16 + e
__global__ void k_pack(const float* __restrict__ Wf, const float* __restrict__ Ws,
                       unsigned short* __restrict__ pw) {
    int idx = blockIdx.x * 256 + threadIdx.x;
    if (idx >= 2 * PACK_HALF) return;
    int e   = idx & 15;
    int l   = (idx >> 4) & 31;
    int nc  = (idx >> 9) & 7;
    int t   = idx >> 12;          // 0..23
    int kc  = t % KC;
    int sel = t / KC;
    int k   = kc * 32 + ((l < 16) ? 0 : 16) + e;
    int n   = nc * 16 + (l & 15);
    const float* W = sel ? Ws : Wf;
    pw[idx] = f2bf(W[k * D + n]);
}

// ------------------------------------------------------- edge GEMM + scatter
__global__ void __launch_bounds__(128)
k_edge_gemm(const unsigned short* __restrict__ xbf, const float* __restrict__ ea,
            const int* __restrict__ esrc, const int* __restrict__ etgt,
            const unsigned short* __restrict__ pw,
            const float* __restrict__ biasF, const float* __restrict__ biasS,
            float* __restrict__ agg) {
    __shared__ __align__(16) unsigned short zlds[TILES * 16 * ZSTRIDE];
    __shared__ int sidx[TILES * 16];
    __shared__ int tidxs[TILES * 16];

    const int  tid = threadIdx.x;
    const long e0  = (long)blockIdx.x * (TILES * 16);

    if (tid < TILES * 16) {
        sidx[tid]  = esrc[e0 + tid];
        tidxs[tid] = etgt[e0 + tid];
    }
    __syncthreads();

    // ---- gather x[src] | x[tgt] (already bf16): pure copy into LDS.
    // 32 rows * 2 segs * 16 chunks of 16B = 1024 chunks / 128 threads = 8 each.
#if defined(USE_ASYNC_LDS)
    #pragma unroll
    for (int w = 0; w < 8; ++w) {
        int c    = w * 128 + tid;
        int part = c & 15;              // 16B chunk within 256B row
        int seg  = (c >> 4) & 1;        // 0 = src, 1 = tgt
        int row  = c >> 5;              // 0..31
        long nrow = seg ? (long)tidxs[row] : (long)sidx[row];
        const unsigned short* gp = xbf + nrow * D + part * 8;
        unsigned short*       lp = &zlds[row * ZSTRIDE + seg * 128 + part * 8];
        __builtin_amdgcn_global_load_async_to_lds_b128(
            (AS1 v4i*)gp, (AS3 v4i*)lp, 0, 0);
    }
#else
    #pragma unroll
    for (int w = 0; w < 8; ++w) {
        int c    = w * 128 + tid;
        int part = c & 15;
        int seg  = (c >> 4) & 1;
        int row  = c >> 5;
        long nrow = seg ? (long)tidxs[row] : (long)sidx[row];
        uint4 v = *(const uint4*)(xbf + nrow * D + part * 8);
        *(uint4*)&zlds[row * ZSTRIDE + seg * 128 + part * 8] = v;
    }
#endif

    // ---- gather edge_attr (f32 -> bf16 in flight): 32 rows * 64 pairs / 128 thr
    #pragma unroll 4
    for (int w = 0; w < 16; ++w) {
        int idx  = w * 128 + tid;
        int pair = idx & 63;
        int row  = idx >> 6;            // 0..31
        float2 v = *(const float2*)(ea + (e0 + row) * D + pair * 2);
        unsigned int u = (unsigned int)f2bf(v.x) | ((unsigned int)f2bf(v.y) << 16);
        *(unsigned int*)&zlds[row * ZSTRIDE + 256 + pair * 2] = u;
    }

#if defined(USE_ASYNC_LDS)
    wait_async_zero();
#endif
    __syncthreads();

    const int wave   = tid >> 5;
    const int lane   = tid & 31;
    const int laneM  = lane & 15;
    const int laneHi = lane >> 4;           // 0 or 1
    const int nc0    = wave;                // cols [16*w, 16*w+16)
    const int nc1    = wave + 4;            // cols [16*w+64, ...)

    v8f accF0[TILES] = {}, accF1[TILES] = {}, accS0[TILES] = {}, accS1[TILES] = {};

    for (int kc = 0; kc < KC; ++kc) {
        const unsigned short* pb = pw + kc * (NC * 512) + lane * 16;
        v16bf bF0 = *(const v16bf*)(pb + nc0 * 512);
        v16bf bF1 = *(const v16bf*)(pb + nc1 * 512);
        v16bf bS0 = *(const v16bf*)(pb + PACK_HALF + nc0 * 512);
        v16bf bS1 = *(const v16bf*)(pb + PACK_HALF + nc1 * 512);
        #pragma unroll
        for (int t = 0; t < TILES; ++t) {
            // A fragment (bf16 16x32): lanes 0-15 hold K 0-15; lanes 16-31 K 16-31
            union { unsigned int u[8]; v16bf v; } a;
            const unsigned short* zr =
                &zlds[(t * 16 + laneM) * ZSTRIDE + kc * 32 + laneHi * 16];
            #pragma unroll
            for (int vg = 0; vg < 8; ++vg) {
                int koff = vg << 1;
                a.u[vg] = *(const unsigned int*)(zr + koff);
            }
            accF0[t] = __builtin_amdgcn_wmma_f32_16x16x32_bf16(false, a.v, false, bF0,
                                                               (short)0, accF0[t], false, false);
            accF1[t] = __builtin_amdgcn_wmma_f32_16x16x32_bf16(false, a.v, false, bF1,
                                                               (short)0, accF1[t], false, false);
            accS0[t] = __builtin_amdgcn_wmma_f32_16x16x32_bf16(false, a.v, false, bS0,
                                                               (short)0, accS0[t], false, false);
            accS1[t] = __builtin_amdgcn_wmma_f32_16x16x32_bf16(false, a.v, false, bS1,
                                                               (short)0, accS1[t], false, false);
        }
    }

    // epilogue: bias + sigmoid(F)*softplus(S), scatter-add to agg[src]
    const int n0 = nc0 * 16 + laneM;
    const int n1 = nc1 * 16 + laneM;
    const float bf0 = biasF[n0], bf1 = biasF[n1];
    const float bs0 = biasS[n0], bs1 = biasS[n1];

    #pragma unroll
    for (int t = 0; t < TILES; ++t) {
        #pragma unroll
        for (int i = 0; i < 8; ++i) {
            int m   = i + laneHi * 8;                 // C layout: hi lanes hold M=8..15
            int src = sidx[t * 16 + m];
            float* dst = agg + (long)src * D;
            float m0 = sigmoid_(accF0[t][i] + bf0) * softplus_(accS0[t][i] + bs0);
            float m1 = sigmoid_(accF1[t][i] + bf1) * softplus_(accS1[t][i] + bs1);
            atomicAdd(dst + n0, m0);
            atomicAdd(dst + n1, m1);
        }
    }
}

// ------------------------------------------------------------- column stats
__global__ void k_stats(const float* __restrict__ agg, float* __restrict__ stats, int N) {
    int col = threadIdx.x;                  // 128 threads: one per column
    int r0  = blockIdx.x * 256;
    int r1  = r0 + 256; if (r1 > N) r1 = N;
    float s = 0.0f, sq = 0.0f;
    for (int r = r0; r < r1; ++r) {
        float v = agg[(long)r * D + col];
        s += v; sq += v * v;
    }
    atomicAdd(&stats[col],     s);
    atomicAdd(&stats[D + col], sq);
}

// --------------------------------------------------- batchnorm + softplus out
__global__ void k_final(const float* __restrict__ x, const float* __restrict__ agg,
                        const float* __restrict__ stats,
                        const float* __restrict__ gamma, const float* __restrict__ beta,
                        float* __restrict__ out, int total, float invN) {
    int idx = blockIdx.x * 256 + threadIdx.x;
    if (idx >= total) return;
    int col = idx & (D - 1);
    float mean = stats[col] * invN;
    float var  = stats[D + col] * invN - mean * mean;
    float v = (agg[idx] - mean) * rsqrtf(var + 1e-5f) * gamma[col] + beta[col];
    out[idx] = softplus_(x[idx] + v);
}

extern "C" void kernel_launch(void* const* d_in, const int* in_sizes, int n_in,
                              void* d_out, int out_size, void* d_ws, size_t ws_size,
                              hipStream_t stream) {
    const float* x     = (const float*)d_in[0];
    const float* ea    = (const float*)d_in[1];
    const int*   esrc  = (const int*)d_in[2];
    const int*   etgt  = (const int*)d_in[3];
    const float* Wf    = (const float*)d_in[4];
    const float* bf    = (const float*)d_in[5];
    const float* Ws    = (const float*)d_in[6];
    const float* bs    = (const float*)d_in[7];
    const float* gamma = (const float*)d_in[8];
    const float* beta  = (const float*)d_in[9];
    float*       out   = (float*)d_out;

    const int N = in_sizes[0] / D;
    const int E = in_sizes[2];

    unsigned short* pw  = (unsigned short*)d_ws;
    unsigned short* xbf = pw + (size_t)2 * PACK_HALF;               // N*D bf16
    float* agg   = (float*)((char*)(xbf) + (size_t)N * D * 2);
    float* stats = agg + (size_t)N * D;

    const int zn = N * D + 2 * D;
    hipLaunchKernelGGL(k_zero, dim3((zn + 255) / 256), dim3(256), 0, stream, agg, zn);
    hipLaunchKernelGGL(k_pack, dim3((2 * PACK_HALF + 255) / 256), dim3(256), 0, stream,
                       Wf, Ws, pw);
    hipLaunchKernelGGL(k_cvt_x, dim3((N * D + 255) / 256), dim3(256), 0, stream,
                       x, xbf, N * D);
    hipLaunchKernelGGL(k_edge_gemm, dim3(E / (TILES * 16)), dim3(128), 0, stream,
                       xbf, ea, esrc, etgt, pw, bf, bs, agg);
    hipLaunchKernelGGL(k_stats, dim3((N + 255) / 256), dim3(128), 0, stream,
                       agg, stats, N);
    hipLaunchKernelGGL(k_final, dim3((N * D + 255) / 256), dim3(256), 0, stream,
                       x, agg, stats, gamma, beta, out, N * D, 1.0f / (float)N);
}
